// SNN_14834817040441
// MI455X (gfx1250) — matrix operand order repaired and checked by reading, not compiled, not run on hardware
//
#include <hip/hip_runtime.h>

typedef __attribute__((ext_vector_type(16))) _Float16 v16h;
typedef __attribute__((ext_vector_type(8)))  float    v8f;
typedef __attribute__((ext_vector_type(2)))  float    f32x2;
typedef __attribute__((ext_vector_type(4)))  int      v4i;

#define DIN  784
#define DH   100
#define DOUT 10
#define BB   8192
#define TT   32
#define NPAD 112   // 7 * 16 (padded hidden dim)
#define KPAD 800   // 25 * 32 (padded input dim)
#define NT   7     // n-tiles of 16
#define KT   25    // k-tiles of 32

#define AS1 __attribute__((address_space(1)))
#define AS3 __attribute__((address_space(3)))

#if defined(__gfx1250__) && \
    __has_builtin(__builtin_amdgcn_global_load_async_to_lds_b128) && \
    __has_builtin(__builtin_amdgcn_s_wait_asynccnt)
#define USE_ASYNC_LDS 1
#else
#define USE_ASYNC_LDS 0
#endif

// ---------------------------------------------------------------------------
// Prep: W1 [DH][DIN] f32 (h-major) -> w1t [KPAD][NPAD] f16
// transposed + zero padded; a lane's B column is read from LDS at a stride of NPAD halves.
// ---------------------------------------------------------------------------
__global__ void snn_prep_w1(const float* __restrict__ W1, _Float16* __restrict__ w1t) {
  int idx = blockIdx.x * blockDim.x + threadIdx.x;
  if (idx >= KPAD * NPAD) return;
  int k = idx / NPAD;
  int n = idx % NPAD;
  float v = (k < DIN && n < DH) ? W1[n * DIN + k] : 0.0f;
  w1t[idx] = (_Float16)v;
}

// ---------------------------------------------------------------------------
// Main SNN kernel.
// Block = 256 threads = 8 wave32; each wave owns 16 batch rows.
// Grid = 8192 / (8*16) = 64 workgroups.
// Membrane potential v lives in the 7 WMMA f32 accumulator tiles per wave.
// logits = (sum_t spike_t) @ W2^T, so we only carry per-neuron spike counts.
// ---------------------------------------------------------------------------
__global__ void __launch_bounds__(256)
snn_main(const float* __restrict__ x,       // [BB][DIN]
         const float* __restrict__ noise,   // [TT][BB][DIN]
         const _Float16* __restrict__ w1t,  // [KPAD][NPAD] f16
         const float* __restrict__ W2,      // [DOUT][DH]
         float* __restrict__ out)           // [BB][DOUT]
{
  extern __shared__ char smem[];                 // KPAD*NPAD*2 = 179200 B (< 320KB/WGP)
  _Float16* lw1 = (_Float16*)smem;

  // Stage the full f16 W1^T into LDS once per workgroup.
#if USE_ASYNC_LDS
  {
    // CDNA5 async copy: GLOBAL_LOAD_ASYNC_TO_LDS_B128, tracked by ASYNCcnt.
    char* gsrc = (char*)w1t;                     // cast away const for builtin
    char* ldst = (char*)smem;
    const int total = (KPAD * NPAD * 2) / 16;    // 11200 x 16B
    for (int i = threadIdx.x; i < total; i += 256) {
      __builtin_amdgcn_global_load_async_to_lds_b128(
          (AS1 v4i*)(gsrc + (size_t)i * 16),
          (AS3 v4i*)(ldst + (size_t)i * 16),
          /*imm offset=*/0, /*cpol=*/0);
    }
    __builtin_amdgcn_s_wait_asynccnt(0);
  }
#else
  {
    const int4* src = (const int4*)w1t;
    int4* dst = (int4*)smem;
    const int total = (KPAD * NPAD * 2) / 16;    // 11200 x 16B
    for (int i = threadIdx.x; i < total; i += 256) dst[i] = src[i];
  }
#endif
  __syncthreads();

  const int lane    = threadIdx.x & 31;
  const int wave    = threadIdx.x >> 5;
  const int rowbase = (blockIdx.x * 8 + wave) * 16;
  const int m       = lane & 15;                 // A-matrix row / C-matrix column
  const int kofs    = (lane < 16) ? 0 : 8;       // 16-bit A-layout half-wave K offset
  const int brow    = rowbase + m;               // this lane's batch row for A

  v8f acc[NT];   // membrane potential tiles (C/D of WMMA)
  v8f cnt[NT];   // accumulated spike counts
  for (int n = 0; n < NT; ++n) { v8f z = {}; acc[n] = z; cnt[n] = z; }

  const float* xrow = x + (size_t)brow * DIN;

  for (int t = 0; t < TT; ++t) {
    const float* nrow = noise + ((size_t)t * BB + (size_t)brow) * DIN;

    for (int kt = 0; kt < KT; ++kt) {
      const int kb = kt * 32;

      // Prefetch next timestep's noise line for this lane's row
      // (speculative global_prefetch_b8; noise streams once at 822 MB).
      if (t + 1 < TT) __builtin_prefetch(nrow + (size_t)BB * DIN + kb, 0, 0);

      // Build A fragment: Poisson spikes (noise < x) packed as f16 {0,1}
      // directly in the ISA 16x32 f16 A layout. Noise is read with
      // non-temporal hint (streamed once; keep L2 for x / W1 reuse).
      union { unsigned u[8]; v16h v; } af;
#pragma unroll
      for (int j = 0; j < 8; ++j) {
        int k = kb + ((j & 4) << 2) + kofs + ((j & 3) << 1);  // +16 for j>=4
        unsigned r = 0u;
        if (k < DIN) {  // padded K region contributes 0 (W1 pad is 0 anyway)
          f32x2 nv = __builtin_nontemporal_load((const f32x2*)(nrow + k));
          f32x2 xv = *(const f32x2*)(xrow + k);
          r = (nv.x < xv.x ? 0x00003C00u : 0u) | (nv.y < xv.y ? 0x3C000000u : 0u);
        }
        af.u[j] = r;
      }

      // B fragments from LDS: lane = N column (lane%16), elements over K in A's table order.
      const _Float16* lrow = lw1 + (size_t)kb * NPAD + m;
#pragma unroll
      for (int nt = 0; nt < NT; ++nt) {
        union { int4 q[2]; v16h v; } bf;
#pragma unroll
        for (int s = 0; s < 16; ++s)
          bf.v[s] = lrow[(size_t)(((s & 8) << 1) + kofs + (s & 7)) * NPAD + nt * 16];
        // v += spikes_t x W1^T   (integrate)
        acc[nt] = __builtin_amdgcn_wmma_f32_16x16x32_f16(
            false, af.v, false, bf.v, (short)0, acc[nt], false, false);
      }
    }

    // Fire + soft reset + count (IF neuron, v_th = 1, soft reset v -= spike).
#pragma unroll
    for (int nt = 0; nt < NT; ++nt) {
#pragma unroll
      for (int r = 0; r < 8; ++r) {
        float v = acc[nt][r];
        float s = (v > 1.0f) ? 1.0f : 0.0f;
        acc[nt][r] = v - s;
        cnt[nt][r] += s;
      }
    }
  }

  // ---- logits = cnt @ W2^T ------------------------------------------------
  // Spill per-wave spike counts to LDS (reusing the W1 region), then each
  // lane computes a few (row, out) dot products of length 100.
  __syncthreads();   // everyone done reading lw1
  float* cl = (float*)smem + (size_t)wave * 16 * NPAD;
#pragma unroll
  for (int nt = 0; nt < NT; ++nt) {
    int n = nt * 16 + m;                 // C layout: N = lane % 16
#pragma unroll
    for (int r = 0; r < 8; ++r) {
      int mm = r + ((lane >> 4) << 3);   // C layout: M = r + 8*(lane/16)
      cl[mm * NPAD + n] = cnt[nt][r];
    }
  }
  __syncthreads();

  for (int idx = lane; idx < 16 * DOUT; idx += 32) {
    int bl = idx / DOUT;
    int o  = idx % DOUT;
    const float* w2r = W2 + o * DH;
    const float* cr  = cl + bl * NPAD;
    float s = 0.0f;
#pragma unroll 4
    for (int h = 0; h < DH; ++h) s += cr[h] * w2r[h];
    out[(size_t)(rowbase + bl) * DOUT + o] = s;
  }
}

// ---------------------------------------------------------------------------
extern "C" void kernel_launch(void* const* d_in, const int* in_sizes, int n_in,
                              void* d_out, int out_size, void* d_ws, size_t ws_size,
                              hipStream_t stream) {
  const float* x     = (const float*)d_in[0];   // [8192][784]
  const float* noise = (const float*)d_in[1];   // [32][8192][784]
  const float* W1    = (const float*)d_in[2];   // [100][784]
  const float* W2    = (const float*)d_in[3];   // [10][100]
  float* out = (float*)d_out;                   // [8192][10]

  _Float16* w1t = (_Float16*)d_ws;              // KPAD*NPAD*2 = 179200 B scratch

  snn_prep_w1<<<(KPAD * NPAD + 255) / 256, 256, 0, stream>>>(W1, w1t);

  const int lds_bytes = KPAD * NPAD * 2;        // 179200 B dynamic LDS
  snn_main<<<BB / 128, 256, lds_bytes, stream>>>(x, noise, w1t, W2, out);
}
